// Seq2Seq_47760036331854
// MI455X (gfx1250) — matrix operand order repaired and checked by reading, not compiled, not run on hardware
//
#include <hip/hip_runtime.h>

#define VOCAB 16384
#define HID   512
#define BATCH 32
#define SEQ   128
#define MAXL  128

typedef __attribute__((ext_vector_type(16))) __bf16 v16bf;
typedef __attribute__((ext_vector_type(8)))  float  v8f;

// ---------------- helpers ----------------

__device__ __forceinline__ unsigned short f2bf(float f) {
    unsigned u = __float_as_uint(f);
    unsigned r = ((u >> 16) & 1u) + 0x7FFFu;   // round to nearest even
    return (unsigned short)((u + r) >> 16);
}

// ---- A fragment: 16x32 (MxK) bf16, row-major source, ld = HID ----
// ISA layout: lanes 0-15 -> M=lane, K pairs {0,1},{2,3},{4,5},{6,7},{16,17}..{22,23}
//             lanes 16-31 -> same M, K shifted by +8 per 16-block.
__device__ __forceinline__ v16bf load_a_frag_g(const unsigned short* __restrict__ src,
                                               int row0, int k0) {
    int lane = threadIdx.x & 31;
    int half = lane >> 4, m = lane & 15;
    union { v16bf v; unsigned int u[8]; } f;
    const unsigned short* p = src + (size_t)(row0 + m) * HID + k0 + 8 * half;
#pragma unroll
    for (int i = 0; i < 8; ++i) {
        int k = (i < 4) ? (2 * i) : (16 + 2 * (i - 4));
        f.u[i] = *(const unsigned int*)(p + k);
    }
    return f.v;
}

template <int NELEM>
__device__ __forceinline__ v16bf load_a_frag_lds(const unsigned short (&src)[NELEM],
                                                 int row0, int k0) {
    int lane = threadIdx.x & 31;
    int half = lane >> 4, m = lane & 15;
    union { v16bf v; unsigned int u[8]; } f;
    int base = (row0 + m) * HID + k0 + 8 * half;
#pragma unroll
    for (int i = 0; i < 8; ++i) {
        int k = (i < 4) ? (2 * i) : (16 + 2 * (i - 4));
        f.u[i] = *(const unsigned int*)&src[base + k];
    }
    return f.v;
}

// ---- B fragment: 32x16 (KxN), B[k][n] = W[n0+n][k0+k], W row-major ld = HID ----
// ISA layout: lanes 0-15 -> N=lane, K pairs {0,1}..{6,7},{16,17}..{22,23};
//             lanes 16-31 -> N=lane-16, K shifted by +8 per 16-block.
__device__ __forceinline__ v16bf load_b_frag_g(const unsigned short* __restrict__ W,
                                               int n0, int k0) {
    int lane = threadIdx.x & 31;
    int half = lane >> 4, n = lane & 15;
    union { v16bf v; unsigned int u[8]; } f;
    const unsigned short* p = W + (size_t)(n0 + n) * HID + k0 + 8 * half;
#pragma unroll
    for (int i = 0; i < 8; ++i) f.u[i] = *(const unsigned int*)(p + ((i < 4) ? (2 * i) : (16 + 2 * (i - 4))));
    return f.v;
}

template <int NELEM>
__device__ __forceinline__ v16bf load_b_frag_lds(const unsigned short (&W)[NELEM],
                                                 int n0, int k0) {
    int lane = threadIdx.x & 31;
    int half = lane >> 4, n = lane & 15;
    union { v16bf v; unsigned int u[8]; } f;
    int base = (n0 + n) * HID + k0 + 8 * half;
#pragma unroll
    for (int i = 0; i < 8; ++i) f.u[i] = *(const unsigned int*)&W[base + ((i < 4) ? (2 * i) : (16 + 2 * (i - 4)))];
    return f.v;
}

__device__ __forceinline__ v8f wmma_bf16(v16bf a, v16bf b, v8f c) {
    return __builtin_amdgcn_wmma_f32_16x16x32_bf16(false, a, false, b,
                                                   (short)0, c, false, false);
}

__device__ __forceinline__ void grid_sync(unsigned* counter, unsigned nwg) {
    __syncthreads();
    if (threadIdx.x == 0) {
        __threadfence();
        unsigned old = atomicAdd(counter, 1u);
        unsigned target = (old / nwg + 1u) * nwg;
        while (__hip_atomic_load(counter, __ATOMIC_ACQUIRE,
                                 __HIP_MEMORY_SCOPE_AGENT) < target) {
            __builtin_amdgcn_s_sleep(2);
        }
        __threadfence();
    }
    __syncthreads();
}

// One GRU step for a 16-column slice [cj, cj+16) of the hidden state.
// Block = 256 threads (8 waves). Waves 0..5 -> (gate g = w>>1, M-half = w&1).
// Weight slices pre-staged in LDS: rows g*16+n hold W[g*HID+cj+n][*].
__device__ __forceinline__ void gru_wg_step(
        const unsigned short (&x_lds)[BATCH * HID],
        const unsigned short (&wih_lds)[48 * HID],
        const unsigned short (&whh_lds)[48 * HID],
        const float* __restrict__ bih, const float* __restrict__ bhh,
        const unsigned short* __restrict__ hbf_cur,
        const float* __restrict__ h_cur,
        float* __restrict__ h_nxt, unsigned short* __restrict__ hbf_nxt,
        float (&gate_lds)[4 * 32 * 16],
        int cj) {
    int wave = threadIdx.x >> 5;
    int lane = threadIdx.x & 31;
    if (wave < 6) {
        int g  = wave >> 1;
        int mh = wave & 1;
        v8f ai = {}, ah = {};
#pragma unroll
        for (int kk = 0; kk < HID; kk += 32) {
            v16bf ax  = load_a_frag_lds(x_lds, mh * 16, kk);
            v16bf bw  = load_b_frag_lds(wih_lds, g * 16, kk);
            ai = wmma_bf16(ax, bw, ai);
            v16bf ah2 = load_a_frag_g(hbf_cur, mh * 16, kk);
            v16bf bh  = load_b_frag_lds(whh_lds, g * 16, kk);
            ah = wmma_bf16(ah2, bh, ah);
        }
        int half = lane >> 4, c = lane & 15;
        float bi  = bih[g * HID + cj + c];
        float bh2 = bhh[g * HID + cj + c];
#pragma unroll
        for (int r = 0; r < 8; ++r) {
            int row = mh * 16 + half * 8 + r;
            float vi = ai[r] + bi, vh = ah[r] + bh2;
            if (g == 0) {
                gate_lds[0 * 512 + row * 16 + c] = 1.f / (1.f + __expf(-(vi + vh)));
            } else if (g == 1) {
                gate_lds[1 * 512 + row * 16 + c] = 1.f / (1.f + __expf(-(vi + vh)));
            } else {
                gate_lds[2 * 512 + row * 16 + c] = vi;   // inn
                gate_lds[3 * 512 + row * 16 + c] = vh;   // hn
            }
        }
    }
    __syncthreads();
    for (int idx = threadIdx.x; idx < 512; idx += 256) {
        int row = idx >> 4, c = idx & 15;
        float r = gate_lds[0 * 512 + idx];
        float z = gate_lds[1 * 512 + idx];
        float n = tanhf(gate_lds[2 * 512 + idx] + r * gate_lds[3 * 512 + idx]);
        float ho = h_cur[row * HID + cj + c];
        float hn = (1.f - z) * n + z * ho;
        h_nxt[row * HID + cj + c] = hn;
        hbf_nxt[row * HID + cj + c] = f2bf(hn);
    }
    __syncthreads();
}

// Stage 16-column GRU weight slices (both matrices) into LDS, converting to bf16.
__device__ __forceinline__ void stage_gru_weights(
        unsigned short (&wih_lds)[48 * HID], unsigned short (&whh_lds)[48 * HID],
        const float* __restrict__ Wih, const float* __restrict__ Whh, int cj) {
    for (int i = threadIdx.x; i < 48 * HID; i += 256) {
        int rr = i >> 9, k = i & 511;
        int g = rr >> 4, n = rr & 15;
        size_t src = (size_t)(g * HID + cj + n) * HID + k;
        wih_lds[i] = f2bf(Wih[src]);
        whh_lds[i] = f2bf(Whh[src]);
    }
}

// ---------------- kernels ----------------

__global__ void convert_bf16_kernel(const float* __restrict__ src,
                                    unsigned short* __restrict__ dst, int n) {
    int i = blockIdx.x * blockDim.x + threadIdx.x;
    if (i < n) dst[i] = f2bf(src[i]);
}

__global__ __launch_bounds__(256) void encoder_kernel(
        const int* __restrict__ input, const float* __restrict__ emb,
        const float* __restrict__ Wih, const float* __restrict__ Whh,
        const float* __restrict__ bih, const float* __restrict__ bhh,
        float* h, unsigned short* hbf, unsigned* counter) {
    __shared__ __align__(16) unsigned short x_lds[BATCH * HID];
    __shared__ __align__(16) unsigned short wih_lds[48 * HID];
    __shared__ __align__(16) unsigned short whh_lds[48 * HID];
    __shared__ float gate_lds[4 * 32 * 16];
    int cj = blockIdx.x * 16;
    stage_gru_weights(wih_lds, whh_lds, Wih, Whh, cj);
    __syncthreads();
    int cur = 0;
    for (int t = 0; t < SEQ; ++t) {
        for (int i = threadIdx.x; i < BATCH * HID; i += 256) {
            int b = i >> 9, k = i & 511;
            int tok = input[b * SEQ + t];
            x_lds[i] = f2bf(emb[(size_t)tok * HID + k]);
        }
        __syncthreads();
        int nxt = cur ^ 1;
        gru_wg_step(x_lds, wih_lds, whh_lds, bih, bhh,
                    hbf + cur * BATCH * HID, h + cur * BATCH * HID,
                    h + nxt * BATCH * HID, hbf + nxt * BATCH * HID,
                    gate_lds, cj);
        cur = nxt;
        grid_sync(counter, 32);
    }
}

__global__ __launch_bounds__(256) void decoder_kernel(
        const float* __restrict__ dec_emb,
        const float* __restrict__ Wih, const float* __restrict__ Whh,
        const float* __restrict__ bih, const float* __restrict__ bhh,
        const unsigned short* __restrict__ outW_bf,
        const float* __restrict__ out_b,
        float* h, unsigned short* hbf,
        unsigned long long* amax, float* out, float* dec_hidden_out,
        unsigned* counter) {
    __shared__ __align__(16) unsigned short x_lds[BATCH * HID];
    __shared__ __align__(16) unsigned short wih_lds[48 * HID];
    __shared__ __align__(16) unsigned short whh_lds[48 * HID];
    __shared__ float gate_lds[4 * 32 * 16];
    const unsigned NWG = 128;
    int wg = blockIdx.x;
    if (wg < 32) {
        stage_gru_weights(wih_lds, whh_lds, Wih, Whh, wg * 16);
        __syncthreads();
    }
    // loop-invariant logits-tile setup
    int wave = threadIdx.x >> 5, lane = threadIdx.x & 31;
    int half = lane >> 4, nn = lane & 15;
    int n0 = (wg * 8 + wave) * 16;
    float ob = out_b[n0 + nn];

    int cur = 0;
    for (int t = 0; t < MAXL; ++t) {
        int nxt = cur ^ 1;
        // ---- GRU phase (WGs 0..31) ----
        if (wg < 32) {
            for (int i = threadIdx.x; i < BATCH * HID; i += 256) {
                int b = i >> 9, k = i & 511;
                int tok = 0;                                   // SOS
                if (t > 0) tok = (int)(~(unsigned)amax[(t - 1) * BATCH + b]) & (VOCAB - 1);
                x_lds[i] = f2bf(dec_emb[(size_t)tok * HID + k]);
            }
            __syncthreads();
            gru_wg_step(x_lds, wih_lds, whh_lds, bih, bhh,
                        hbf + cur * BATCH * HID, h + cur * BATCH * HID,
                        h + nxt * BATCH * HID, hbf + nxt * BATCH * HID,
                        gate_lds, wg * 16);
        }
        grid_sync(counter, NWG);
        // ---- logits GEMM + argmax (128 WGs x 8 waves = 1024 N-tiles) ----
        const unsigned short* hsrc = hbf + nxt * BATCH * HID;
#pragma unroll
        for (int mh = 0; mh < 2; ++mh) {
            v8f acc = {};
#pragma unroll
            for (int kk = 0; kk < HID; kk += 32) {
                v16bf a = load_a_frag_g(hsrc, mh * 16, kk);
                v16bf b = load_b_frag_g(outW_bf, n0, kk);
                acc = wmma_bf16(a, b, acc);
            }
#pragma unroll
            for (int r = 0; r < 8; ++r) {
                int brow = mh * 16 + half * 8 + r;
                float Lg = acc[r] + ob;
                out[((size_t)brow * MAXL + t) * VOCAB + n0 + nn] = Lg;
                float v = Lg; int vi = n0 + nn;
                for (int d = 1; d < 16; d <<= 1) {           // 16 lanes share a row
                    float ov = __shfl_xor(v, d);
                    int   oi = __shfl_xor(vi, d);
                    if (ov > v || (ov == v && oi < vi)) { v = ov; vi = oi; }
                }
                if (nn == 0) {
                    unsigned m = __float_as_uint(v);
                    m = (m & 0x80000000u) ? ~m : (m | 0x80000000u);
                    unsigned long long p =
                        ((unsigned long long)m << 32) | (unsigned)(~vi);
                    atomicMax(&amax[t * BATCH + brow], p);
                }
            }
        }
        cur = nxt;
        grid_sync(counter, NWG);
    }
    if (wg == 0) {
        for (int i = threadIdx.x; i < BATCH * HID; i += 256)
            dec_hidden_out[i] = h[cur * BATCH * HID + i];
    }
}

__global__ __launch_bounds__(256) void logsoftmax_kernel(float* __restrict__ out) {
    __shared__ float sm[8], ss[8], s_lse;
    float* p = out + (size_t)blockIdx.x * VOCAB;
    float m = -INFINITY, s = 0.f;
    for (int i = threadIdx.x; i < VOCAB; i += 256) {
        float x = p[i];
        float nm = fmaxf(m, x);
        s = s * __expf(m - nm) + __expf(x - nm);
        m = nm;
    }
    for (int d = 1; d < 32; d <<= 1) {                 // wave32 reduce
        float om = __shfl_xor(m, d);
        float os = __shfl_xor(s, d);
        float nm = fmaxf(m, om);
        s = s * __expf(m - nm) + os * __expf(om - nm);
        m = nm;
    }
    int wave = threadIdx.x >> 5;
    if ((threadIdx.x & 31) == 0) { sm[wave] = m; ss[wave] = s; }
    __syncthreads();
    if (threadIdx.x == 0) {
        float M = sm[0], S = ss[0];
        for (int w = 1; w < 8; ++w) {
            float nm = fmaxf(M, sm[w]);
            S = S * __expf(M - nm) + ss[w] * __expf(sm[w] - nm);
            M = nm;
        }
        s_lse = M + logf(S);
    }
    __syncthreads();
    float lse = s_lse;
    for (int i = threadIdx.x; i < VOCAB; i += 256) p[i] -= lse;
}

// ---------------- host ----------------

extern "C" void kernel_launch(void* const* d_in, const int* in_sizes, int n_in,
                              void* d_out, int out_size, void* d_ws, size_t ws_size,
                              hipStream_t stream) {
    const int*   input   = (const int*)d_in[0];
    const float* enc_emb = (const float*)d_in[1];
    const float* enc_Wih = (const float*)d_in[2];
    const float* enc_Whh = (const float*)d_in[3];
    const float* enc_bih = (const float*)d_in[4];
    const float* enc_bhh = (const float*)d_in[5];
    const float* dec_emb = (const float*)d_in[6];
    const float* dec_Wih = (const float*)d_in[7];
    const float* dec_Whh = (const float*)d_in[8];
    const float* dec_bih = (const float*)d_in[9];
    const float* dec_bhh = (const float*)d_in[10];
    const float* out_W   = (const float*)d_in[11];
    const float* out_b   = (const float*)d_in[12];

    // workspace layout (offsets 256B aligned)
    const size_t o_outW = 0;                                       // VOCAB*HID bf16
    const size_t o_h    = o_outW + (size_t)VOCAB * HID * 2;        // 2*B*H f32
    const size_t o_hbf  = o_h    + (size_t)2 * BATCH * HID * 4;    // 2*B*H bf16
    const size_t o_amax = o_hbf  + (size_t)2 * BATCH * HID * 2;    // 128*32 u64
    const size_t o_cnt  = o_amax + (size_t)MAXL * BATCH * 8;       // counters
    const size_t total  = o_cnt + 256;
    if (ws_size < total) return;

    char* ws = (char*)d_ws;
    unsigned short*     outW_bf = (unsigned short*)(ws + o_outW);
    float*              hbuf    = (float*)(ws + o_h);
    unsigned short*     hbf     = (unsigned short*)(ws + o_hbf);
    unsigned long long* amax    = (unsigned long long*)(ws + o_amax);
    unsigned*           cnt_enc = (unsigned*)(ws + o_cnt);
    unsigned*           cnt_dec = (unsigned*)(ws + o_cnt + 128);

    // zero h ping-pong, hbf, argmax, counters in one async memset
    hipMemsetAsync(ws + o_h, 0, total - o_h, stream);

    // out_W -> bf16 copy (lives in L2 for the 128 logits GEMMs)
    {
        int n = VOCAB * HID;
        convert_bf16_kernel<<<(n + 255) / 256, 256, 0, stream>>>(out_W, outW_bf, n);
    }

    float* log_probs  = (float*)d_out;                     // [B, MAXL, VOCAB]
    float* dec_hidden = log_probs + (size_t)BATCH * MAXL * VOCAB;

    encoder_kernel<<<32, 256, 0, stream>>>(input, enc_emb, enc_Wih, enc_Whh,
                                           enc_bih, enc_bhh, hbuf, hbf, cnt_enc);

    decoder_kernel<<<128, 256, 0, stream>>>(dec_emb, dec_Wih, dec_Whh,
                                            dec_bih, dec_bhh, outW_bf, out_b,
                                            hbuf, hbf, amax, log_probs,
                                            dec_hidden, cnt_dec);

    logsoftmax_kernel<<<BATCH * MAXL, 256, 0, stream>>>(log_probs);
}
